// MambaBlock_60473139528515
// MI455X (gfx1250) — matrix operand order repaired and checked by reading, not compiled, not run on hardware
//
#include <hip/hip_runtime.h>
#include <hip/hip_bf16.h>
#include <math.h>

typedef __attribute__((ext_vector_type(16))) _Float16 v16h;
typedef __attribute__((ext_vector_type(8)))  _Float16 v8h;
typedef __attribute__((ext_vector_type(8)))  float    v8f;

#define DIM   1024
#define LSEQ  1024
#define NBAT  2
#define DI    2048            // d_inner
#define E2    4096            // 2*d_inner
#define RNK   64              // DT_RANK
#define NST   16              // D_STATE
#define NRC   96              // DT_RANK + 2*D_STATE
#define NTOK  (NBAT*LSEQ)     // 2048 tokens

__device__ __forceinline__ float sigmoidf_(float x) { return 1.0f / (1.0f + __expf(-x)); }

// ---------------- f32 -> f16 convert ----------------
__global__ void cvt_f16_kernel(const float* __restrict__ in, _Float16* __restrict__ out, int n) {
  int i = blockIdx.x * blockDim.x + threadIdx.x;
  if (i < n) out[i] = (_Float16)in[i];
}

// ---------------- LayerNorm (one block per token row) ----------------
__global__ __launch_bounds__(256) void ln_kernel(const float* __restrict__ x,
                                                 const float* __restrict__ w,
                                                 const float* __restrict__ b,
                                                 _Float16* __restrict__ xn) {
  __shared__ float s1[256], s2[256];
  int row = blockIdx.x, tid = threadIdx.x;
  const float* xr = x + (size_t)row * DIM;
  float a = 0.f, q = 0.f;
  for (int i = tid; i < DIM; i += 256) { float v = xr[i]; a += v; q += v * v; }
  s1[tid] = a; s2[tid] = q; __syncthreads();
  for (int st = 128; st > 0; st >>= 1) {
    if (tid < st) { s1[tid] += s1[tid + st]; s2[tid] += s2[tid + st]; }
    __syncthreads();
  }
  float mu  = s1[0] * (1.0f / DIM);
  float var = s2[0] * (1.0f / DIM) - mu * mu;
  float inv = rsqrtf(var + 1e-5f);
  for (int i = tid; i < DIM; i += 256)
    xn[(size_t)row * DIM + i] = (_Float16)((xr[i] - mu) * inv * w[i] + b[i]);
}

// ---------------- NT GEMM via V_WMMA_F32_16X16X32_F16 ----------------
// C[m][n] = sum_k A[m][k]*B[n][k];  A: MxK f16 row-major, B: NxK f16 row-major, C: MxN f32.
// Optional residual: C = A*B^T + Res (same MxN f32 layout).
// CONTRACT: M % 128 == 0 (true for all call sites: M = 2048). N may be ragged.
// Block = 256 threads = 8 wave32s, wave grid 2(M) x 4(N); wave tile 64x32 (4x2 WMMA tiles);
// block tile 128x128. 8 WMMAs per 12 b128 loads per K-step.
// NOTE: addresses are per-iteration GEPs off the kernarg pointers so clang infers the
// global address space (global_load_b128 / global_prefetch_b8, not flat_*).
__global__ __launch_bounds__(256) void gemm_nt_wmma(const _Float16* __restrict__ A,
                                                    const _Float16* __restrict__ B,
                                                    float* __restrict__ C,
                                                    const float* __restrict__ Res,
                                                    int M, int N, int K) {
  int lane = threadIdx.x & 31;
  int wave = threadIdx.x >> 5;
  int wm = wave & 1;            // 2 waves along M
  int wn = wave >> 1;           // 4 waves along N
  int m0 = blockIdx.y * 128 + wm * 64;
  int n0 = blockIdx.x * 128 + wn * 32;
  int lr = lane & 15;
  int hi = lane >> 4;

  v8f acc[4][2] = {};
  size_t arow[4], brow[2];
#pragma unroll
  for (int i = 0; i < 4; ++i) {
    int m = m0 + i * 16 + lr;                    // in range by contract (M % 128 == 0)
    arow[i] = (size_t)m * K + hi * 16;           // A-frag: elems 0..15 -> K=kb2..kb2+15
  }
#pragma unroll
  for (int j = 0; j < 2; ++j) {
    int n = n0 + j * 16 + lr; if (n >= N) n = N - 1;   // clamp (ragged N); garbage masked at store
    brow[j] = (size_t)n * K + hi * 16;           // B-frag: elems 0..15 -> K=kb2..kb2+15
  }

  for (int k0 = 0; k0 < K; k0 += 32) {
    v16h af[4], bf[2];
#pragma unroll
    for (int i = 0; i < 4; ++i) {
      const _Float16* ap = A + arow[i] + k0;
      union { v16h v; v8h h[2]; } u;
      u.h[0] = *(const v8h*)(ap);
      u.h[1] = *(const v8h*)(ap + 8);
      af[i] = u.v;
      __builtin_prefetch(ap + 64, 0, 3);         // near-scope prefetch, next cacheline of A stream
    }
#pragma unroll
    for (int j = 0; j < 2; ++j) {
      const _Float16* bp = B + brow[j] + k0;
      bf[j] = *(const v16h*)(bp);
      __builtin_prefetch(bp + 64, 0, 3);         // next cacheline of B stream
    }
#pragma unroll
    for (int i = 0; i < 4; ++i)
#pragma unroll
      for (int j = 0; j < 2; ++j)
        acc[i][j] = __builtin_amdgcn_wmma_f32_16x16x32_f16(
            false, af[i], false, bf[j], (short)0, acc[i][j], false, false);
  }

  // Epilogue: D elem r -> (M = mb + r, N = n0 + j*16 + lr). M always in range; guard N only.
#pragma unroll
  for (int i = 0; i < 4; ++i) {
    int mb = m0 + i * 16 + hi * 8;
#pragma unroll
    for (int j = 0; j < 2; ++j) {
      int n = n0 + j * 16 + lr;
      if (n < N) {
        size_t o = (size_t)mb * N + n;
        if (Res) {
#pragma unroll
          for (int r = 0; r < 8; ++r) { C[o] = acc[i][j][r] + Res[o]; o += N; }
        } else {
#pragma unroll
          for (int r = 0; r < 8; ++r) { C[o] = acc[i][j][r]; o += N; }
        }
      }
    }
  }
}

// ---------------- depthwise causal conv (D_CONV=4) + SiLU ----------------
__global__ void conv_silu_kernel(const float* __restrict__ xz, const float* __restrict__ cw,
                                 const float* __restrict__ cb, float* __restrict__ uf,
                                 _Float16* __restrict__ uh) {
  int idx = blockIdx.x * blockDim.x + threadIdx.x;    // over NBAT*LSEQ*DI
  if (idx >= NBAT * LSEQ * DI) return;
  int e = idx % DI;
  int l = (idx / DI) % LSEQ;
  int b = idx / (DI * LSEQ);
  float acc = cb[e];
  size_t base = (size_t)(b * LSEQ) * E2 + e;          // xs = xz[..., :DI]
#pragma unroll
  for (int k = 0; k < 4; ++k) {
    int ll = l - 3 + k;
    if (ll >= 0) acc += cw[e * 4 + k] * xz[base + (size_t)ll * E2];
  }
  float u = acc * sigmoidf_(acc);
  uf[idx] = u;
  uh[idx] = (_Float16)u;
}

// ---------------- split dt_r out of x_dbl (as f16 for dt GEMM) ----------------
__global__ void split_dtr_kernel(const float* __restrict__ xdbl, _Float16* __restrict__ dtr) {
  int i = blockIdx.x * blockDim.x + threadIdx.x;      // NTOK*RNK
  if (i < NTOK * RNK) {
    int t = i / RNK, r = i % RNK;
    dtr[i] = (_Float16)xdbl[(size_t)t * NRC + r];
  }
}

// ---------------- selective scan: one thread per (b, e) channel ----------------
__global__ __launch_bounds__(256) void scan_kernel(
    const float* __restrict__ dtlin, const float* __restrict__ bdt,
    const float* __restrict__ alog,  const float* __restrict__ dskip,
    const float* __restrict__ uf,    const float* __restrict__ xdbl,
    const float* __restrict__ xz,    _Float16* __restrict__ yh) {
  int t = blockIdx.x * blockDim.x + threadIdx.x;      // 0..NBAT*DI-1
  if (t >= NBAT * DI) return;
  int b = t / DI, e = t % DI;
  float Ae[NST], h[NST];
#pragma unroll
  for (int n = 0; n < NST; ++n) { Ae[n] = -__expf(alog[e * NST + n]); h[n] = 0.f; }
  float bd = bdt[e], ds = dskip[e];
  for (int l = 0; l < LSEQ; ++l) {
    int tok = b * LSEQ + l;
    float dl = dtlin[(size_t)tok * DI + e] + bd;
    float dt = (dl > 20.f) ? dl : log1pf(__expf(dl));        // softplus
    float uu = uf[(size_t)tok * DI + e];
    float du = dt * uu;
    const float* Bp = xdbl + (size_t)tok * NRC + RNK;        // B_ssm (wave-uniform)
    const float* Cp = Bp + NST;                              // C_ssm
    float y = 0.f;
#pragma unroll
    for (int n = 0; n < NST; ++n) {
      float dA = __expf(dt * Ae[n]);
      h[n] = dA * h[n] + du * Bp[n];
      y += h[n] * Cp[n];
    }
    float zz = xz[(size_t)tok * E2 + DI + e];
    float yv = (y + uu * ds) * (zz * sigmoidf_(zz));          // * silu(z)
    yh[(size_t)tok * DI + e] = (_Float16)yv;
  }
}

extern "C" void kernel_launch(void* const* d_in, const int* in_sizes, int n_in,
                              void* d_out, int out_size, void* d_ws, size_t ws_size,
                              hipStream_t stream) {
  const float* x       = (const float*)d_in[0];
  const float* ln_w    = (const float*)d_in[1];
  const float* ln_b    = (const float*)d_in[2];
  const float* w_in    = (const float*)d_in[3];
  const float* conv_w  = (const float*)d_in[4];
  const float* conv_b  = (const float*)d_in[5];
  const float* w_xproj = (const float*)d_in[6];
  const float* w_dt    = (const float*)d_in[7];
  const float* b_dt    = (const float*)d_in[8];
  const float* a_log   = (const float*)d_in[9];
  const float* d_skip  = (const float*)d_in[10];
  const float* w_out   = (const float*)d_in[11];
  float* out = (float*)d_out;

  char* ws = (char*)d_ws; size_t off = 0;
  auto carve = [&](size_t bytes) { void* p = ws + off; off += (bytes + 255) & ~(size_t)255; return p; };
  _Float16* w_in_h  = (_Float16*)carve((size_t)E2  * DIM * 2);
  _Float16* w_xp_h  = (_Float16*)carve((size_t)NRC * DI  * 2);
  _Float16* w_dt_h  = (_Float16*)carve((size_t)DI  * RNK * 2);
  _Float16* w_out_h = (_Float16*)carve((size_t)DIM * DI  * 2);
  _Float16* xn_h    = (_Float16*)carve((size_t)NTOK * DIM * 2);
  float*    xzbuf   = (float*)   carve((size_t)NTOK * E2  * 4);
  float*    u_f     = (float*)   carve((size_t)NTOK * DI  * 4);
  _Float16* u_h     = (_Float16*)carve((size_t)NTOK * DI  * 2);
  float*    xdbl    = (float*)   carve((size_t)NTOK * NRC * 4);
  _Float16* dtr_h   = (_Float16*)carve((size_t)NTOK * RNK * 2);
  float*    dtlin   = (float*)   carve((size_t)NTOK * DI  * 4);
  _Float16* y_h     = (_Float16*)carve((size_t)NTOK * DI  * 2);
  (void)ws_size; (void)in_sizes; (void)n_in; (void)out_size;

  dim3 blk(256);
  // weight conversion f32 -> f16
  cvt_f16_kernel<<<(E2 * DIM + 255) / 256, blk, 0, stream>>>(w_in,    w_in_h,  E2 * DIM);
  cvt_f16_kernel<<<(NRC * DI + 255) / 256, blk, 0, stream>>>(w_xproj, w_xp_h,  NRC * DI);
  cvt_f16_kernel<<<(DI * RNK + 255) / 256, blk, 0, stream>>>(w_dt,    w_dt_h,  DI * RNK);
  cvt_f16_kernel<<<(DIM * DI + 255) / 256, blk, 0, stream>>>(w_out,   w_out_h, DIM * DI);
  // layernorm
  ln_kernel<<<NTOK, blk, 0, stream>>>(x, ln_w, ln_b, xn_h);
  // xz = xn @ w_in^T : 2048 x 4096 x 1024
  gemm_nt_wmma<<<dim3(E2 / 128, NTOK / 128), blk, 0, stream>>>(xn_h, w_in_h, xzbuf, nullptr, NTOK, E2, DIM);
  // u = silu(causal_conv(xs))
  conv_silu_kernel<<<(NTOK * DI) / 256, blk, 0, stream>>>(xzbuf, conv_w, conv_b, u_f, u_h);
  // x_dbl = u @ w_xproj^T : 2048 x 96 x 2048  (ragged N)
  gemm_nt_wmma<<<dim3(1, NTOK / 128), blk, 0, stream>>>(u_h, w_xp_h, xdbl, nullptr, NTOK, NRC, DI);
  split_dtr_kernel<<<(NTOK * RNK) / 256, blk, 0, stream>>>(xdbl, dtr_h);
  // dt_lin = dt_r @ w_dt^T : 2048 x 2048 x 64
  gemm_nt_wmma<<<dim3(DI / 128, NTOK / 128), blk, 0, stream>>>(dtr_h, w_dt_h, dtlin, nullptr, NTOK, DI, RNK);
  // selective scan + gating
  scan_kernel<<<(NBAT * DI) / 256, blk, 0, stream>>>(dtlin, b_dt, a_log, d_skip, u_f, xdbl, xzbuf, y_h);
  // out = y @ w_out^T + residual : 2048 x 1024 x 2048, fused residual epilogue
  gemm_nt_wmma<<<dim3(DIM / 128, NTOK / 128), blk, 0, stream>>>(y_h, w_out_h, out, x, NTOK, DIM, DI);
}
